// Ranked_Attention_57062935494770
// MI455X (gfx1250) — compile-verified, matrix operand order checked by reading, not run on hardware
//
#include <hip/hip_runtime.h>
#include <hip/hip_bf16.h>

// ---------------------------------------------------------------------------
// Ranked attention on gfx1250 (MI455X), f16 WMMA (16x16x32), f32 accumulate.
//   D = 300 (padded to 320 = 10 K-tiles of 32), L = 160 (10 M-tiles of 16)
//   One workgroup (256 threads = 8 wave32) per batch element.
//   Dual independent accumulator chains per wave to hide WMMA latency.
// ---------------------------------------------------------------------------

typedef __attribute__((ext_vector_type(16))) _Float16 v16h;
typedef __attribute__((ext_vector_type(8)))  _Float16 v8h;
typedef __attribute__((ext_vector_type(4)))  _Float16 v4h;
typedef __attribute__((ext_vector_type(8)))  float    v8f;

#define LSEQ 160
#define DDIM 300
#define DPAD 320
#define LD   (LSEQ * DDIM)         // 48000
#define NI_TILES 20                // DPAD / 16
#define KT_TILES 10                // DPAD / 32
#define MI_TILES 10                // LSEQ / 16
#define PK_ELEMS (NI_TILES * KT_TILES * 32 * 16)   // 102400 f16 per matrix

__device__ __forceinline__ float sigmoidf_(float x) {
  return 1.f / (1.f + __expf(-x));
}

// A-operand (and LDS-resident B-operand) load: 16-bit 16x32 tile.
// lane<16:  row m=lane,     K = kt*32 + {0..7, 16..23}
// lane>=16: row m=lane-16,  K = kt*32 + {8..15, 24..31}
__device__ __forceinline__ v16h lds_tile16(const _Float16* base, int t16, int kt, int lane) {
  const int m  = lane & 15;
  const int kh = lane >> 4;
  const _Float16* p = base + (t16 * 16 + m) * DPAD + kt * 32 + kh * 8;
  v8h lo = *(const v8h*)(p);
  v8h hi = *(const v8h*)(p + 16);
  return __builtin_shufflevector(lo, hi, 0,1,2,3,4,5,6,7,8,9,10,11,12,13,14,15);
}

// ---------------------------------------------------------------------------
// Prepack: f32 weight (o rows, d cols, 300x300) -> f16 B-operand tiles.
// Flat element r = ni*5120 + kt*512 + lane*16 + e matches the main kernel's
// 32-byte-per-lane v16h load.
// ---------------------------------------------------------------------------
__global__ void prepack_weights(const float* __restrict__ Wproj,
                                const float* __restrict__ WqCD,
                                const float* __restrict__ WkCD,
                                _Float16* __restrict__ pk) {
  int idx = blockIdx.x * blockDim.x + threadIdx.x;
  if (idx >= 3 * PK_ELEMS) return;
  const int mat = idx / PK_ELEMS;
  const int r   = idx % PK_ELEMS;
  const int e    = r & 15;
  const int lane = (r >> 4) & 31;
  const int kt   = (r >> 9) % KT_TILES;
  const int ni   = r / (KT_TILES * 512);
  const int n  = ni * 16 + (lane & 15);
  const int kh = lane >> 4;
  const int k  = kt * 32 + kh * 8 + (e < 8 ? e : e + 8);
  const float* W = (mat == 0) ? Wproj : (mat == 1) ? WqCD : WkCD;
  float v = (n < DDIM && k < DDIM) ? W[n * DDIM + k] : 0.f;
  pk[idx] = (_Float16)v;
}

// Stage a (160,300) f32 row-major feature block into (160,320) f16 LDS, zero pad.
__device__ __forceinline__ void stage_feat(const float* __restrict__ src,
                                           _Float16* dst, int tid) {
  for (int i4 = tid; i4 < LSEQ * 75; i4 += 256) {      // 75 float4 per row
    const int row = i4 / 75;
    const int c4  = i4 - row * 75;
    const float4 v = *(const float4*)(src + row * DDIM + c4 * 4);
    v4h h = { (_Float16)v.x, (_Float16)v.y, (_Float16)v.z, (_Float16)v.w };
    *(v4h*)(dst + row * DPAD + c4 * 4) = h;
  }
  for (int p = tid; p < LSEQ * (DPAD - DDIM); p += 256) {
    const int row = p / (DPAD - DDIM);
    dst[row * DPAD + DDIM + (p - row * (DPAD - DDIM))] = (_Float16)0.f;
  }
}

// Tile epilogue: bias add, f16 -> LDS (unconditional), f32 -> global (one
// predicated block, not per-element, to avoid exec-mask churn).
__device__ __forceinline__ void tile_epilogue(const v8f& acc, float bcol,
                                              int row0, int col, bool wr,
                                              float* __restrict__ outG,
                                              _Float16* sOut) {
  float vals[8];
#pragma unroll
  for (int v = 0; v < 8; ++v) vals[v] = acc[v] + bcol;
#pragma unroll
  for (int v = 0; v < 8; ++v)
    sOut[(row0 + v) * DPAD + col] = (_Float16)vals[v];   // padded cols store 0
  if (wr) {
#pragma unroll
    for (int v = 0; v < 8; ++v)
      outG[(size_t)(row0 + v) * DDIM + col] = vals[v];
  }
}

// out(160x320) = A(160x320, LDS f16) @ B + bias. B prepacked in global (f16).
// Two M-tiles per pass share the preloaded B registers -> 2 independent
// WMMA accumulator chains per wave.
__device__ __forceinline__ void gemm_320(const _Float16* sA,
                                         const _Float16* __restrict__ pkB,
                                         const float* __restrict__ bias,
                                         float* __restrict__ outG,
                                         _Float16* sOut,
                                         int lane, int wave) {
  for (int ni = wave; ni < NI_TILES; ni += 8) {
    v16h Bt[KT_TILES];
#pragma unroll
    for (int kt = 0; kt < KT_TILES; ++kt)
      Bt[kt] = *(const v16h*)(pkB + (size_t)(((ni * KT_TILES + kt) * 32) + lane) * 16);
    const int col = ni * 16 + (lane & 15);
    const float bcol = (col < DDIM) ? bias[col] : 0.f;
    const int r0 = (lane >> 4) * 8;
    const bool wr = (outG != nullptr) && (col < DDIM);
    for (int mi = 0; mi < MI_TILES; mi += 2) {
      v8f acc0 = {0.f, 0.f, 0.f, 0.f, 0.f, 0.f, 0.f, 0.f};
      v8f acc1 = {0.f, 0.f, 0.f, 0.f, 0.f, 0.f, 0.f, 0.f};
#pragma unroll
      for (int kt = 0; kt < KT_TILES; ++kt) {
        v16h a0 = lds_tile16(sA, mi, kt, lane);
        v16h a1 = lds_tile16(sA, mi + 1, kt, lane);
        acc0 = __builtin_amdgcn_wmma_f32_16x16x32_f16(false, a0, false, Bt[kt],
                                                      (short)0, acc0, false, false);
        acc1 = __builtin_amdgcn_wmma_f32_16x16x32_f16(false, a1, false, Bt[kt],
                                                      (short)0, acc1, false, false);
      }
      tile_epilogue(acc0, bcol, mi * 16 + r0, col, wr, outG, sOut);
      tile_epilogue(acc1, (mi + 1) * 16 + r0 < LSEQ ? bcol : bcol, // same bias
                    (mi + 1) * 16 + r0, col, wr, outG, sOut);
    }
  }
}

// ---------------------------------------------------------------------------
// Main kernel: one block per batch element.
// ---------------------------------------------------------------------------
__global__ __launch_bounds__(256) void ranked_attention_main(
    const float* __restrict__ T_feat, const float* __restrict__ CD_feat,
    const float* __restrict__ bproj,
    const float* __restrict__ WqT_w, const float* __restrict__ WqT_b,
    const float* __restrict__ WkT_w, const float* __restrict__ WkT_b,
    const float* __restrict__ bqCD, const float* __restrict__ bkCD,
    const float* __restrict__ W3w,
    const _Float16* __restrict__ pkProj, const _Float16* __restrict__ pkWq,
    const _Float16* __restrict__ pkWk,
    float* __restrict__ outP, float* __restrict__ outH) {
  constexpr float INV_SQRT_D = 0.05773502691896258f;  // 1/sqrt(300)

  extern __shared__ char smem_raw[];
  _Float16* sR0 = (_Float16*)smem_raw;        // 160x320 f16 (100KB)
  _Float16* sR1 = sR0 + LSEQ * DPAD;          // 100KB
  _Float16* sR2 = sR1 + LSEQ * DPAD;          // 100KB
  float* sMean  = (float*)(sR2 + LSEQ * DPAD);
  float* sQt    = sMean + DPAD;
  float* sWv    = sQt + DPAD;
  float* sZW    = sWv + DPAD;                 // ZT[j]*W3w[j]
  float* sLin   = sZW + LSEQ;
  float* sAlpha = sLin + LSEQ;
  float* sScal  = sAlpha + LSEQ;              // 4 floats

  const int b    = blockIdx.x;
  const int tid  = threadIdx.x;
  const int lane = tid & 31;
  const int wave = tid >> 5;

  // ---- Phase 1: proj_T = T @ Wproj^T + b  (f32 -> d_out, f16 -> sR1) ----
  stage_feat(T_feat + (size_t)b * LD, sR0, tid);
  __syncthreads();
  gemm_320(sR0, pkProj, bproj, outP + (size_t)b * LD, sR1, lane, wave);
  __syncthreads();

  // ---- Phase 2: ZT path, algebraically reduced:
  //   m = mean_i projT[i];  qT = WqT m + bqT;  w = WkT^T qT;  c = qT.bkT
  //   ZT[j] = sigmoid((projT[j].w + c)/sqrt(D));  sZW[j] = ZT[j]*W3w[j]
  for (int d = tid; d < DDIM; d += 256) {
    float s = 0.f;
    for (int i = 0; i < LSEQ; ++i) s += (float)sR1[i * DPAD + d];
    sMean[d] = s * (1.f / (float)LSEQ);
  }
  __syncthreads();
  for (int o = tid; o < DDIM; o += 256) {
    float s = WqT_b[o];
    const float* wr = WqT_w + o * DDIM;
    for (int d = 0; d < DDIM; ++d) s += wr[d] * sMean[d];
    sQt[o] = s;
  }
  __syncthreads();
  for (int d = tid; d < DDIM; d += 256) {
    float s = 0.f;
    for (int o = 0; o < DDIM; ++o) s += WkT_w[o * DDIM + d] * sQt[o];
    sWv[d] = s;
  }
  if (tid == 0) {
    float c = 0.f;
    for (int o = 0; o < DDIM; ++o) c += sQt[o] * WkT_b[o];
    sScal[0] = c;
  }
  __syncthreads();
  for (int j = tid; j < LSEQ; j += 256) {
    float s = sScal[0];
    for (int d = 0; d < DDIM; ++d) s += (float)sR1[j * DPAD + d] * sWv[d];
    sZW[j] = sigmoidf_(s * INV_SQRT_D) * W3w[j];
  }

  // ---- Phase 3: proj_CD (f16 -> sR1; overwrites proj_T after ZT done) ----
  stage_feat(CD_feat + (size_t)b * LD, sR0, tid);
  __syncthreads();                       // orders ZT reads before sR1 rewrite
  gemm_320(sR0, pkProj, bproj, nullptr, sR1, lane, wave);
  __syncthreads();

  // ---- Phase 4: qCD -> sR0, kCD -> sR2 (both read sR1; independent) ----
  gemm_320(sR1, pkWq, bqCD, nullptr, sR0, lane, wave);
  gemm_320(sR1, pkWk, bkCD, nullptr, sR2, lane, wave);
  for (int i = tid; i < LSEQ; i += 256) sLin[i] = 0.f;
  __syncthreads();

  // ---- Phase 5: alphaCD = qCD @ kCD^T; fused sigmoid * ZT[j] * W3w[j];
  //      row-reduce into lin[i] (softmax over i; W3_b is a uniform shift).
  //      Two N-tiles per pass share the A load -> 2 independent chains. ----
  for (int t = wave; t < MI_TILES * (MI_TILES / 2); t += 8) {   // 50 pair-tiles
    const int mi  = t / (MI_TILES / 2);
    const int nj0 = (t - mi * (MI_TILES / 2)) * 2;
    v8f acc0 = {0.f, 0.f, 0.f, 0.f, 0.f, 0.f, 0.f, 0.f};
    v8f acc1 = {0.f, 0.f, 0.f, 0.f, 0.f, 0.f, 0.f, 0.f};
#pragma unroll
    for (int kt = 0; kt < KT_TILES; ++kt) {
      v16h a  = lds_tile16(sR0, mi, kt, lane);
      v16h b0 = lds_tile16(sR2, nj0, kt, lane);
      v16h b1 = lds_tile16(sR2, nj0 + 1, kt, lane);
      acc0 = __builtin_amdgcn_wmma_f32_16x16x32_f16(false, a, false, b0,
                                                    (short)0, acc0, false, false);
      acc1 = __builtin_amdgcn_wmma_f32_16x16x32_f16(false, a, false, b1,
                                                    (short)0, acc1, false, false);
    }
    const float zw0 = sZW[nj0 * 16 + (lane & 15)];
    const float zw1 = sZW[(nj0 + 1) * 16 + (lane & 15)];
    const int row0 = mi * 16 + (lane >> 4) * 8;
#pragma unroll
    for (int v = 0; v < 8; ++v) {
      float c0 = sigmoidf_(acc0[v] * INV_SQRT_D) * zw0;
      float c1 = sigmoidf_(acc1[v] * INV_SQRT_D) * zw1;
      float c = c0 + c1;
      c += __shfl_xor(c, 1);
      c += __shfl_xor(c, 2);
      c += __shfl_xor(c, 4);
      c += __shfl_xor(c, 8);             // sum over the 16 columns in this half
      if ((lane & 15) == 0) atomicAdd(&sLin[row0 + v], c);
    }
  }
  __syncthreads();

  // ---- Phase 6: softmax over rows, HCD = alpha_f[i] * proj_CD[i,:] ----
  if (tid == 0) {
    float mx = -3.4e38f;
    for (int i = 0; i < LSEQ; ++i) mx = fmaxf(mx, sLin[i]);
    sScal[1] = mx;
  }
  __syncthreads();
  for (int i = tid; i < LSEQ; i += 256) sAlpha[i] = __expf(sLin[i] - sScal[1]);
  __syncthreads();
  if (tid == 0) {
    float s = 0.f;
    for (int i = 0; i < LSEQ; ++i) s += sAlpha[i];
    sScal[2] = 1.f / s;
  }
  __syncthreads();
  float* oH = outH + (size_t)b * LD;
  const float inv = sScal[2];
  for (int i4 = tid; i4 < LSEQ * 75; i4 += 256) {
    const int row = i4 / 75;
    const int c4  = i4 - row * 75;
    const float a = sAlpha[row] * inv;
    const v4h h = *(const v4h*)(sR1 + row * DPAD + c4 * 4);
    float4 o;
    o.x = a * (float)h[0];
    o.y = a * (float)h[1];
    o.z = a * (float)h[2];
    o.w = a * (float)h[3];
    *(float4*)(oH + row * DDIM + c4 * 4) = o;
  }
}

extern "C" void kernel_launch(void* const* d_in, const int* in_sizes, int n_in,
                              void* d_out, int out_size, void* d_ws, size_t ws_size,
                              hipStream_t stream) {
  (void)n_in; (void)out_size; (void)ws_size;
  const float* T_feat  = (const float*)d_in[0];
  const float* CD_feat = (const float*)d_in[1];
  const float* Wproj   = (const float*)d_in[2];
  const float* bproj   = (const float*)d_in[3];
  const float* WqT_w   = (const float*)d_in[4];
  const float* WqT_b   = (const float*)d_in[5];
  const float* WqCD_w  = (const float*)d_in[6];
  const float* WqCD_b  = (const float*)d_in[7];
  const float* WkT_w   = (const float*)d_in[8];
  const float* WkT_b   = (const float*)d_in[9];
  const float* WkCD_w  = (const float*)d_in[10];
  const float* WkCD_b  = (const float*)d_in[11];
  const float* W3_w    = (const float*)d_in[12];
  // W3_b (d_in[13]) is a uniform pre-softmax shift: mathematically a no-op.

  const int B = in_sizes[0] / LD;                    // 1024
  _Float16* pk = (_Float16*)d_ws;                    // needs 3*PK_ELEMS*2 = 600KB

  // 1) Prepack f16 B-operand weight tiles (L2-resident, shared by all batches).
  {
    const int total = 3 * PK_ELEMS;
    prepack_weights<<<(total + 255) / 256, 256, 0, stream>>>(Wproj, WqCD_w, WkCD_w, pk);
  }

  // 2) Main fused kernel: one workgroup per batch, ~313 KB dynamic LDS.
  float* outP = (float*)d_out;
  float* outH = outP + (size_t)B * LD;
  const size_t smem = (size_t)3 * LSEQ * DPAD * sizeof(_Float16)
                    + (size_t)(3 * DPAD + 3 * LSEQ + 4) * sizeof(float);
  ranked_attention_main<<<B, 256, smem, stream>>>(
      T_feat, CD_feat, bproj, WqT_w, WqT_b, WkT_w, WkT_b, WqCD_b, WkCD_b, W3_w,
      pk, pk + PK_ELEMS, pk + 2 * PK_ELEMS, outP, outH);
}
